// HopAttentionLayer_62706522522384
// MI455X (gfx1250) — matrix operand order repaired and checked by reading, not compiled, not run on hardware
//
#include <hip/hip_runtime.h>

// HopAttentionLayer for MI455X (gfx1250, wave32).
// Dead-code elimination: reference's hop/temp chain never reaches the output.
// Pipeline: [K1] h = inp@w (WMMA f16->f32), emit h_f32 + hT_f16
//           [K2] s1 = h@a1, s2 = h@a2 (wave-per-row reduction)
//           [K3] fused masked softmax + P@h (WMMA) + elu, P kept in LDS
//                (320 KB/WGP makes a 16x4096 f16 probability tile resident).

typedef _Float16 v16h __attribute__((ext_vector_type(16)));
typedef _Float16 v8h  __attribute__((ext_vector_type(8)));
typedef _Float16 v4h  __attribute__((ext_vector_type(4)));
typedef float    v8f  __attribute__((ext_vector_type(8)));
typedef int      v4i  __attribute__((ext_vector_type(4)));

#define N_NODES 4096
#define F_IN    512
#define F_OUT   256
// P row stride in halves: 4096 + 8 pad halves (16 B) so that consecutive rows
// are 8208 B apart -> 8208 mod 256 = 16 -> the 16 lanes of an A-fragment read
// land on 16 distinct 16 B bank slices: conflict-free ds_load_b128.
#define PROW    (N_NODES + 8)

__device__ __forceinline__ v8f wmma_f16(v16h a, v16h b, v8f c) {
  // (neg_a, A, neg_b, B, c_mod, C, reuse_a, reuse_b)
  return __builtin_amdgcn_wmma_f32_16x16x32_f16(false, a, false, b, (short)0, c,
                                                false, false);
}

// ---------------------------------------------------------------------------
// Kernel 1: h = inp @ w.  One wave per 16x16 output tile, K = 512 in steps
// of 32 via v_wmma_f32_16x16x32_f16.  Writes h row-major f32 (for the score
// vectors) and hT column-major f16 (B-operand-friendly for kernel 3).
// ---------------------------------------------------------------------------
__global__ __launch_bounds__(256) void gemm_h_kernel(
    const float* __restrict__ inp, const float* __restrict__ wmat,
    float* __restrict__ h32, _Float16* __restrict__ hT16) {
  int wave = (blockIdx.x * blockDim.x + threadIdx.x) >> 5;  // 4096 waves
  int lane = threadIdx.x & 31;
  int mt = wave >> 4;             // 0..255
  int nt = wave & 15;             // 0..15
  int m0 = mt * 16, n0 = nt * 16;
  int hi = lane >> 4;             // lane half selects K sub-block
  int l15 = lane & 15;
  int koff_a = hi * 16;           // A: lanes>=16 hold K+16
  int koff_b = hi * 16;           // B: lanes>=16 hold K+16
  const float* arow = inp + (m0 + l15) * F_IN;
  int ncol = n0 + l15;

  v8f c = {};
  for (int kk = 0; kk < F_IN; kk += 32) {
    v16h a, b;
#pragma unroll
    for (int i = 0; i < 8; ++i) {
      a[i]     = (_Float16)arow[kk + koff_a + i];        // K = kk+koff+0..7
      a[i + 8] = (_Float16)arow[kk + koff_a + 8 + i];    // K = kk+koff+8..15
    }
#pragma unroll
    for (int i = 0; i < 16; ++i)                          // coalesced across lanes
      b[i] = (_Float16)wmat[(kk + koff_b + i) * F_OUT + ncol];
    c = wmma_f16(a, b, c);
  }

  int mb = m0 + hi * 8;           // C layout: vgpr v -> M = v (+8 for hi half)
#pragma unroll
  for (int v = 0; v < 8; ++v) {
    int m = mb + v;
    float val = c[v];
    h32[m * F_OUT + ncol] = val;
    hT16[(long)ncol * N_NODES + m] = (_Float16)val;
  }
}

// ---------------------------------------------------------------------------
// Kernel 2: s1[i] = h[i,:].a1, s2[i] = h[i,:].a2.  One wave per row.
// ---------------------------------------------------------------------------
__global__ __launch_bounds__(256) void scores_kernel(
    const float* __restrict__ h32, const float* __restrict__ avec,
    float* __restrict__ s1g, float* __restrict__ s2g) {
  int row  = (blockIdx.x * blockDim.x + threadIdx.x) >> 5;  // 4096 rows
  int lane = threadIdx.x & 31;
  float s1 = 0.f, s2 = 0.f;
#pragma unroll
  for (int i = 0; i < 8; ++i) {
    int n = lane + i * 32;
    float hv = h32[row * F_OUT + n];
    s1 += hv * avec[n];
    s2 += hv * avec[F_OUT + n];
  }
#pragma unroll
  for (int off = 16; off > 0; off >>= 1) {
    s1 += __shfl_xor(s1, off, 32);
    s2 += __shfl_xor(s2, off, 32);
  }
  if (lane == 0) { s1g[row] = s1; s2g[row] = s2; }
}

// ---------------------------------------------------------------------------
// Kernel 3: fused attention.  One block = 16 output rows, 512 threads =
// 16 waves.  Stage 1: wave w owns row w -> masked exp(lrelu(s1+s2)-8) into
// LDS (f16) + row denominator; adj streamed once as b128.  Stage 2: wave w
// owns a 16-col slab of h'; 128 x v_wmma_f32_16x16x32_f16 over K=4096;
// normalize + elu in the epilogue (normalization cancels the -8 shift).
// ---------------------------------------------------------------------------
#define SMEM_P_OFF   16384                       // after 4096 f32 of s2
#define SMEM_SUM_OFF (SMEM_P_OFF + 16 * PROW * 2)
#define SMEM_TOTAL   (SMEM_SUM_OFF + 64)

__global__ __launch_bounds__(512) void attn_kernel(
    const int* __restrict__ adj, const float* __restrict__ s1g,
    const float* __restrict__ s2g, const _Float16* __restrict__ hT16,
    float* __restrict__ out) {
  extern __shared__ __align__(16) char smem[];
  float*    s2l  = (float*)smem;
  _Float16* Pl   = (_Float16*)(smem + SMEM_P_OFF);
  float*    sums = (float*)(smem + SMEM_SUM_OFF);

  int m0   = blockIdx.x * 16;
  int tid  = threadIdx.x;
  int wv   = tid >> 5;            // 0..15
  int lane = tid & 31;

  for (int j = tid; j < N_NODES; j += 512) s2l[j] = s2g[j];
  __syncthreads();

  // ---- stage 1: unnormalized softmax numerators (shift by 8: cancels in
  //      normalization; keeps exp() inside f16 range without a max pass) ----
  int row = m0 + wv;
  float s1r = s1g[row];
  const v4i* adjrow4 = (const v4i*)(adj + (long)row * N_NODES);
  _Float16* prow_w = Pl + wv * PROW;
  float sum = 0.f;
  for (int it = 0; it < 32; ++it) {
    int j = (it * 32 + lane) * 4;          // 32 lanes x 16 B = 512 B contiguous
    v4i av = adjrow4[it * 32 + lane];
    v4h pv;
#pragma unroll
    for (int q = 0; q < 4; ++q) {
      float e = s1r + s2l[j + q];
      e = (e > 0.f) ? e : 0.2f * e;                  // leaky relu, alpha = 0.2
      float p = (av[q] > 0) ? __expf(e - 8.0f) : 0.0f;  // mask -> exact 0
      pv[q] = (_Float16)p;
      sum += p;
    }
    *(v4h*)(prow_w + j) = pv;                        // one b64 DS store
  }
#pragma unroll
  for (int off = 16; off > 0; off >>= 1) sum += __shfl_xor(sum, off, 32);
  if (lane == 0) sums[wv] = sum;
  __syncthreads();

  // ---- stage 2: h'[m0..m0+15, wv*16..wv*16+15] = P @ h ----
  int hi = lane >> 4, l15 = lane & 15;
  int koff_a = hi * 16, koff_b = hi * 16;
  int ncol = wv * 16 + l15;
  const _Float16* hTrow = hT16 + (long)ncol * N_NODES;  // contiguous in K
  const _Float16* prow  = Pl + l15 * PROW;              // A row = output row

  v8f c = {};
  // No unroll cap: the compiler unrolls this deeply (16x in practice),
  // putting many independent ds_load_b128 / 32B global loads in flight
  // ahead of the (D->C accumulate, hazard-free) WMMA chain.
  for (int kk = 0; kk < N_NODES; kk += 32) {
    v8h alo = *(const v8h*)(prow + kk + koff_a);        // K +0..7  (b128, no conflict)
    v8h ahi = *(const v8h*)(prow + kk + koff_a + 8);    // K +8..15
    v16h a;
#pragma unroll
    for (int i = 0; i < 8; ++i) { a[i] = alo[i]; a[i + 8] = ahi[i]; }
    v16h b = *(const v16h*)(hTrow + kk + koff_b);       // one 32 B global load
    c = wmma_f16(a, b, c);
  }

  int mb = hi * 8;
#pragma unroll
  for (int v = 0; v < 8; ++v) {
    int m = mb + v;
    float val = c[v] / sums[m];                         // softmax normalize
    val = (val > 0.f) ? val : (__expf(val) - 1.0f);     // elu
    out[(long)(m0 + m) * F_OUT + ncol] = val;
  }
}

// ---------------------------------------------------------------------------
extern "C" void kernel_launch(void* const* d_in, const int* in_sizes, int n_in,
                              void* d_out, int out_size, void* d_ws,
                              size_t ws_size, hipStream_t stream) {
  const float* inp  = (const float*)d_in[0];   // (4096, 512)
  const int*   adj  = (const int*)d_in[1];     // (4096, 4096)
  const float* wmat = (const float*)d_in[2];   // (512, 256)
  const float* avec = (const float*)d_in[3];   // (512,)

  float* out = (float*)d_out;                  // (4096, 256)

  char* ws = (char*)d_ws;                      // needs ~6.04 MB scratch
  float*    h32  = (float*)ws;                                // 4 MB
  _Float16* hT16 = (_Float16*)(ws + 4u * 1024 * 1024);        // 2 MB
  float*    s1g  = (float*)(ws + 6u * 1024 * 1024);           // 16 KB
  float*    s2g  = (float*)(ws + 6u * 1024 * 1024 + 16384);   // 16 KB

  gemm_h_kernel<<<512, 256, 0, stream>>>(inp, wmat, h32, hT16);
  scores_kernel<<<512, 256, 0, stream>>>(h32, avec, s1g, s2g);
  attn_kernel<<<256, 512, SMEM_TOTAL, stream>>>(adj, s1g, s2g, hT16, out);
}
